// ClustCNNEdgeEncoder_1589137899699
// MI455X (gfx1250) — matrix operand order repaired and checked by reading, not compiled, not run on hardware
//
#include <hip/hip_runtime.h>

typedef __attribute__((ext_vector_type(16))) _Float16 v16h;
typedef __attribute__((ext_vector_type(8)))  _Float16 v8h;
typedef __attribute__((ext_vector_type(8)))  float    v8f;
typedef __attribute__((ext_vector_type(4)))  float    v4f;

union Frag { v16h v; v8h h[2]; };

#define CHUNK 128   // edges per pipeline pass (8 passes cover 1024 edges)

// ---------------------------------------------------------------------------
// Kernel 1: per-cluster bounding box -> {mins[3], ranges[3], gs, pad}
// ---------------------------------------------------------------------------
__global__ void bbox_kernel(const float* __restrict__ data,
                            const int*   __restrict__ clusts,
                            const unsigned char* __restrict__ mask,
                            float* __restrict__ bbox) {
  __shared__ float smin[3][128];
  __shared__ float smax[3][128];
  const int c = blockIdx.x;
  const int t = threadIdx.x;           // 0..127
  const int idx = clusts[c * 128 + t];
  const bool m = mask[c * 128 + t] != 0;
#pragma unroll
  for (int d = 0; d < 3; ++d) {
    const float p = data[idx * 5 + d];
    smin[d][t] = m ? p : 1e9f;
    smax[d][t] = m ? p : -1e9f;
  }
  __syncthreads();
  for (int s = 64; s > 0; s >>= 1) {
    if (t < s) {
#pragma unroll
      for (int d = 0; d < 3; ++d) {
        smin[d][t] = fminf(smin[d][t], smin[d][t + s]);
        smax[d][t] = fmaxf(smax[d][t], smax[d][t + s]);
      }
    }
    __syncthreads();
  }
  if (t == 0) {
    float rmax = 0.f;
#pragma unroll
    for (int d = 0; d < 3; ++d) {
      const float mn = smin[d][0];
      const float rg = smax[d][0] - mn;
      bbox[c * 8 + d]     = mn;
      bbox[c * 8 + 3 + d] = rg;
      rmax = fmaxf(rmax, rg);
    }
    bbox[c * 8 + 6] = 1.0f / (rmax + 1.0f);   // gs
    bbox[c * 8 + 7] = 0.f;
  }
}

// ---------------------------------------------------------------------------
// Kernel 2: voxelize one edge (two clusters) into a 32^3 f16 volume.
// 1024 threads: thread t owns (i = t>>5, j = t&31, k = 0..31) in registers.
// ---------------------------------------------------------------------------
__global__ void vox_kernel(const float* __restrict__ data,
                           const int*   __restrict__ clusts,
                           const unsigned char* __restrict__ mask,
                           const long long* __restrict__ eidx,
                           const float* __restrict__ bbox,
                           _Float16* __restrict__ vox,
                           int chunkBase) {
  __shared__ float ov[128 * 96];   // [point][dim(3)][bin(32)]
  __shared__ float wsh[128];
  const int e = chunkBase + blockIdx.x;
  const int t = threadIdx.x;       // 0..1023
  const int i = t >> 5;
  const int j = t & 31;
  const float ns = 1.0f / 32.0f;

  float acc[32];
#pragma unroll
  for (int k = 0; k < 32; ++k) acc[k] = 0.f;

  for (int side = 0; side < 2; ++side) {
    const int cid = (int)eidx[side * 1024 + e];
    const float gs = bbox[cid * 8 + 6];
    // stage fractional overlaps for all 128 points
    for (int q = t; q < 128 * 96; q += 1024) {
      const int p = q / 96;
      const int r = q - p * 96;
      const int d = r >> 5;
      const int b = r & 31;
      const int idx = clusts[cid * 128 + p];
      const float pt = data[idx * 5 + d];
      const float mn = bbox[cid * 8 + d];
      const float rg = bbox[cid * 8 + 3 + d];
      const float v  = (pt - mn - rg * 0.5f - 0.5f) * gs + 0.5f;
      const float lo = (float)b * ns;
      const float o  = fminf(v + gs, lo + ns) - fmaxf(v, lo);
      ov[q] = fmaxf(o, 0.f) / gs;
    }
    if (t < 128) {
      const int idx = clusts[cid * 128 + t];
      wsh[t] = data[idx * 5 + 4] * (mask[cid * 128 + t] ? 1.0f : 0.0f);
    }
    __syncthreads();
    for (int p = 0; p < 128; ++p) {
      const float a = wsh[p] * ov[p * 96 + i] * ov[p * 96 + 32 + j];
      const v4f* ozv = (const v4f*)&ov[p * 96 + 64];   // 16B-aligned z-row
#pragma unroll
      for (int g = 0; g < 8; ++g) {
        const v4f o4 = ozv[g];
#pragma unroll
        for (int k2 = 0; k2 < 4; ++k2)
          acc[g * 4 + k2] = fmaf(a, o4[k2], acc[g * 4 + k2]);
      }
    }
    __syncthreads();
  }
  v8h* out = (v8h*)(vox + (size_t)blockIdx.x * 32768 + (size_t)t * 32);
#pragma unroll
  for (int g = 0; g < 4; ++g) {
    v8h pk;
#pragma unroll
    for (int k2 = 0; k2 < 8; ++k2) pk[k2] = (_Float16)acc[g * 8 + k2];
    out[g] = pk;
  }
}

// ---------------------------------------------------------------------------
// WMMA 16-bit lane layout helpers: lane = hi*16 + lo16.
// A 16x32 half-runs: t<8 -> k = hi*8+t ; t>=8 -> k = 16+hi*8+(t-8)
// B 32x16 half-runs: same k order as A (a_k)
// ---------------------------------------------------------------------------
__device__ __forceinline__ int a_k(int hi, int t) {
  return (t < 8) ? (hi * 8 + t) : (16 + hi * 8 + (t - 8));
}

// ---------------------------------------------------------------------------
// Kernel 3: conv1  (1->16 ch, 32^3 -> 16^3).  M=4096, N=16, K=27 (pad 32).
// grid = CHUNK*32 blocks x 256 thr; each wave = one 16x16 tile, 1 WMMA.
// K order = tap (dz*9+dy*3+dx).
// ---------------------------------------------------------------------------
__global__ void conv1_kernel(const _Float16* __restrict__ vox,
                             const float* __restrict__ W1,
                             const float* __restrict__ b1,
                             _Float16* __restrict__ c1out) {
  const int blk  = blockIdx.x;
  const int e    = blk >> 5;
  const int wave = threadIdx.x >> 5;
  const int lane = threadIdx.x & 31;
  const int mtile = (blk & 31) * 8 + wave;          // 0..255
  const _Float16* V = vox + (size_t)e * 32768;

  const int lo16 = lane & 15, hi = lane >> 4;
  v16h bfrag;
#pragma unroll
  for (int t = 0; t < 16; ++t) {
    const int k = a_k(hi, t);
    bfrag[t] = (k < 27) ? (_Float16)W1[lo16 * 27 + k] : (_Float16)0.f;
  }
  const int pos = mtile * 16 + lo16;
  const int oz = pos >> 8, oy = (pos >> 4) & 15, ox = pos & 15;
  v16h afrag;
#pragma unroll
  for (int t = 0; t < 16; ++t) {
    const int k = a_k(hi, t);
    _Float16 v = (_Float16)0.f;
    if (k < 27) {
      const int dz = k / 9, dy = (k / 3) % 3, dx = k % 3;
      const int iz = 2 * oz + dz, iy = 2 * oy + dy, ix = 2 * ox + dx;
      if (iz < 32 && iy < 32 && ix < 32) v = V[iz * 1024 + iy * 32 + ix];
    }
    afrag[t] = v;
  }
  v8f c = {};
  c = __builtin_amdgcn_wmma_f32_16x16x32_f16(false, afrag, false, bfrag,
                                             (short)0, c, false, false);
  _Float16* O = c1out + (size_t)e * 65536;
#pragma unroll
  for (int r = 0; r < 8; ++r) {
    const int m = hi * 8 + r;
    const float y = fmaxf(c[r] + b1[lo16], 0.f);
    O[(size_t)(mtile * 16 + m) * 16 + lo16] = (_Float16)y;
  }
}

// ---------------------------------------------------------------------------
// Kernel 4: conv2 (16->32 ch, 16^3 -> 8^3).  M=512, N=32, K=432 (pad 448).
// K order = tap*16 + c  -> A half-runs are 8 contiguous channels (b128 load).
// LDS weights transposed [n][Kpad]  -> B half-runs contiguous (ds b128).
// grid = CHUNK*8 x 256 thr; 14 WMMA per wave.
// ---------------------------------------------------------------------------
__global__ void conv2_kernel(const _Float16* __restrict__ c1,
                             const float* __restrict__ W2,
                             const float* __restrict__ b2,
                             _Float16* __restrict__ c2out) {
  __shared__ _Float16 Wl[32 * 448];   // [n][k] 28KB
  const int blk  = blockIdx.x;
  const int e    = blk >> 3;
  const int wave = threadIdx.x >> 5;
  const int lane = threadIdx.x & 31;
  const int tile = (blk & 7) * 8 + wave;   // 0..63
  const int mtile = tile >> 1;             // 0..31
  const int ntile = tile & 1;              // 0..1

  const _Float16* I = c1 + (size_t)e * 65536;
  __builtin_prefetch(I + (size_t)threadIdx.x * 256, 0, 1);  // 128KB activation tile

  for (int q = threadIdx.x; q < 32 * 448; q += 256) {
    const int n = q / 448, k = q - n * 448;
    _Float16 w = (_Float16)0.f;
    if (k < 432) {
      const int tap = k >> 4, ch = k & 15;          // K = tap*16 + c
      w = (_Float16)W2[n * 432 + ch * 27 + tap];
    }
    Wl[q] = w;
  }
  __syncthreads();

  const int lo16 = lane & 15, hi = lane >> 4;
  const int m = mtile * 16 + lo16;
  const int oz = m >> 6, oy = (m >> 3) & 7, ox = m & 7;
  const _Float16* Brow = &Wl[(ntile * 16 + lo16) * 448];
  v8f c = {};
  for (int s = 0; s < 14; ++s) {
    Frag a, b;
#pragma unroll
    for (int half = 0; half < 2; ++half) {
      const int base = s * 32 + (half ? (16 + hi * 8) : (hi * 8));
      v8h r = {};
      if (base < 432) {
        const int tap = base >> 4, c0 = base & 15;
        const int dz = tap / 9, dy = (tap / 3) % 3, dx = tap % 3;
        const int iz = 2 * oz + dz, iy = 2 * oy + dy, ix = 2 * ox + dx;
        if (iz < 16 && iy < 16 && ix < 16)
          r = *(const v8h*)(I + (size_t)((iz * 16 + iy) * 16 + ix) * 16 + c0);
      }
      a.h[half] = r;
    }
    const _Float16* bp = Brow + s * 32 + hi * 8;
    b.h[0] = *(const v8h*)bp;
    b.h[1] = *(const v8h*)(bp + 16);
    c = __builtin_amdgcn_wmma_f32_16x16x32_f16(false, a.v, false, b.v,
                                               (short)0, c, false, false);
  }
  _Float16* O = c2out + (size_t)e * 16384;
#pragma unroll
  for (int r = 0; r < 8; ++r) {
    const int mm = mtile * 16 + hi * 8 + r;
    const int nn = ntile * 16 + lo16;
    const float y = fmaxf(c[r] + b2[nn], 0.f);
    O[(size_t)mm * 32 + nn] = (_Float16)y;
  }
}

// ---------------------------------------------------------------------------
// Kernel 5: conv3 (32->64 ch, 8^3 -> 4^3).  M=64, N=64, K=864 (pad 896).
// K order = tap*32 + c.  grid = CHUNK*4 x 128 thr; 28 WMMA per wave.
// ---------------------------------------------------------------------------
__global__ void conv3_kernel(const _Float16* __restrict__ c2,
                             const float* __restrict__ W3,
                             const float* __restrict__ b3,
                             _Float16* __restrict__ c3out) {
  __shared__ _Float16 Wl[16 * 896];   // [n-sub][k] 28KB
  const int blk  = blockIdx.x;
  const int e    = blk >> 2;
  const int ntile = blk & 3;
  const int wave = threadIdx.x >> 5;   // 0..3 == mtile
  const int lane = threadIdx.x & 31;

  const _Float16* I = c2 + (size_t)e * 16384;
  __builtin_prefetch(I + (size_t)threadIdx.x * 128, 0, 1);  // 32KB activation tile

  for (int q = threadIdx.x; q < 16 * 896; q += 128) {
    const int n = q / 896, k = q - n * 896;
    _Float16 w = (_Float16)0.f;
    if (k < 864) {
      const int tap = k >> 5, ch = k & 31;          // K = tap*32 + c
      w = (_Float16)W3[(size_t)(ntile * 16 + n) * 864 + ch * 27 + tap];
    }
    Wl[q] = w;
  }
  __syncthreads();

  const int lo16 = lane & 15, hi = lane >> 4;
  const int m = wave * 16 + lo16;       // 0..63
  const int oz = m >> 4, oy = (m >> 2) & 3, ox = m & 3;
  const _Float16* Brow = &Wl[lo16 * 896];
  v8f c = {};
  for (int s = 0; s < 28; ++s) {
    Frag a, b;
#pragma unroll
    for (int half = 0; half < 2; ++half) {
      const int base = s * 32 + (half ? (16 + hi * 8) : (hi * 8));
      v8h r = {};
      if (base < 864) {
        const int tap = base >> 5, c0 = base & 31;
        const int dz = tap / 9, dy = (tap / 3) % 3, dx = tap % 3;
        const int iz = 2 * oz + dz, iy = 2 * oy + dy, ix = 2 * ox + dx;
        if (iz < 8 && iy < 8 && ix < 8)
          r = *(const v8h*)(I + (size_t)((iz * 8 + iy) * 8 + ix) * 32 + c0);
      }
      a.h[half] = r;
    }
    const _Float16* bp = Brow + s * 32 + hi * 8;
    b.h[0] = *(const v8h*)bp;
    b.h[1] = *(const v8h*)(bp + 16);
    c = __builtin_amdgcn_wmma_f32_16x16x32_f16(false, a.v, false, b.v,
                                               (short)0, c, false, false);
  }
  _Float16* O = c3out + (size_t)e * 4096;
#pragma unroll
  for (int r = 0; r < 8; ++r) {
    const int mm = wave * 16 + hi * 8 + r;
    const int nn = ntile * 16 + lo16;
    const float y = fmaxf(c[r] + b3[nn], 0.f);
    O[(size_t)mm * 64 + nn] = (_Float16)y;
  }
}

// ---------------------------------------------------------------------------
// Kernel 6: mean-pool + FC.  GEMM M=128(edges) N=64 K=64 via WMMA (2 steps).
// Pooled matrix + transposed Wfc staged f16 in LDS; vector ds loads.
// grid = 1 block x 256 thr per chunk.
// ---------------------------------------------------------------------------
__global__ void fc_kernel(const _Float16* __restrict__ c3,
                          const float* __restrict__ Wfc,
                          const float* __restrict__ bfc,
                          float* __restrict__ out,
                          int chunkBase) {
  __shared__ _Float16 P[128 * 64];   // pooled [edge][ch], 16KB
  __shared__ _Float16 Wl[64 * 64];   // Wfc^T  [n][k],     8KB
  const int t = threadIdx.x;
  for (int q = t; q < 128 * 64; q += 256) {
    const int le = q >> 6, ch = q & 63;
    const _Float16* I = c3 + (size_t)le * 4096 + ch;
    float s = 0.f;
#pragma unroll
    for (int p = 0; p < 64; ++p) s += (float)I[p * 64];
    P[q] = (_Float16)(s * (1.0f / 64.0f));
  }
  for (int q = t; q < 4096; q += 256) {
    const int n = q >> 6, k = q & 63;
    Wl[q] = (_Float16)Wfc[k * 64 + n];       // transpose to [n][k]
  }
  __syncthreads();

  const int wave = t >> 5, lane = t & 31;
  const int lo16 = lane & 15, hi = lane >> 4;
  for (int tt = 0; tt < 4; ++tt) {
    const int tile = wave * 4 + tt;     // 32 tiles: mtile 0..7 x ntile 0..3
    const int mtile = tile >> 2, ntile = tile & 3;
    const _Float16* Arow = &P[(mtile * 16 + lo16) * 64];
    const _Float16* Brow = &Wl[(ntile * 16 + lo16) * 64];
    v8f c = {};
#pragma unroll
    for (int s = 0; s < 2; ++s) {
      Frag a, b;
      a.h[0] = *(const v8h*)(Arow + s * 32 + hi * 8);
      a.h[1] = *(const v8h*)(Arow + s * 32 + 16 + hi * 8);
      const _Float16* bp = Brow + s * 32 + hi * 8;
      b.h[0] = *(const v8h*)bp;
      b.h[1] = *(const v8h*)(bp + 16);
      c = __builtin_amdgcn_wmma_f32_16x16x32_f16(false, a.v, false, b.v,
                                                 (short)0, c, false, false);
    }
#pragma unroll
    for (int r = 0; r < 8; ++r) {
      const int m = mtile * 16 + hi * 8 + r;      // local edge
      const int n = ntile * 16 + lo16;
      out[(size_t)(chunkBase + m) * 64 + n] = c[r] + bfc[n];
    }
  }
}

// ---------------------------------------------------------------------------
extern "C" void kernel_launch(void* const* d_in, const int* in_sizes, int n_in,
                              void* d_out, int out_size, void* d_ws, size_t ws_size,
                              hipStream_t stream) {
  (void)in_sizes; (void)n_in; (void)out_size; (void)ws_size;
  const float* data = (const float*)d_in[0];
  const int*   clusts = (const int*)d_in[1];
  const unsigned char* mask = (const unsigned char*)d_in[2];
  const long long* eidx = (const long long*)d_in[3];
  const float* W1 = (const float*)d_in[4];
  const float* b1 = (const float*)d_in[5];
  const float* W2 = (const float*)d_in[6];
  const float* b2 = (const float*)d_in[7];
  const float* W3 = (const float*)d_in[8];
  const float* b3 = (const float*)d_in[9];
  const float* Wfc = (const float*)d_in[10];
  const float* bfc = (const float*)d_in[11];
  float* out = (float*)d_out;

  char* w = (char*)d_ws;
  float* bbox = (float*)w;
  size_t off = 8192;
  _Float16* vox = (_Float16*)(w + off); off += (size_t)CHUNK * 32768 * 2; // 8MB
  _Float16* c1  = (_Float16*)(w + off); off += (size_t)CHUNK * 65536 * 2; // 16MB
  _Float16* c2  = (_Float16*)(w + off); off += (size_t)CHUNK * 16384 * 2; // 4MB
  _Float16* c3  = (_Float16*)(w + off); off += (size_t)CHUNK * 4096  * 2; // 1MB

  bbox_kernel<<<256, 128, 0, stream>>>(data, clusts, mask, bbox);
  for (int chunk = 0; chunk < 1024 / CHUNK; ++chunk) {
    const int base = chunk * CHUNK;
    vox_kernel  <<<CHUNK,      1024, 0, stream>>>(data, clusts, mask, eidx, bbox, vox, base);
    conv1_kernel<<<CHUNK * 32, 256,  0, stream>>>(vox, W1, b1, c1);
    conv2_kernel<<<CHUNK * 8,  256,  0, stream>>>(c1, W2, b2, c2);
    conv3_kernel<<<CHUNK * 4,  128,  0, stream>>>(c2, W3, b3, c3);
    fc_kernel   <<<1,          256,  0, stream>>>(c3, Wfc, bfc, out, base);
  }
}
